// LlamaAttentionLESS_45165876084770
// MI455X (gfx1250) — matrix operand order repaired and checked by reading, not compiled, not run on hardware
//
#include <hip/hip_runtime.h>
#include <math.h>

// ---------- types ----------
typedef __bf16 bf16_t;
typedef bf16_t v16bf __attribute__((ext_vector_type(16)));
typedef float  v8f   __attribute__((ext_vector_type(8)));

struct alignas(16) q16 { unsigned int w[4]; };
union Frag { v16bf v; q16 q[2]; };

#define S_LEN 2048
#define D_DIM 4096
#define H_NUM 32
#define HD_DIM 128
#define KD_DIM 64
#define KH_DIM 128
#define HEAVY_N 256
#define RECENT_N 256
#define ATT_SCALE 0.08838834764831845f   // 1/sqrt(128)
// GEMM dims (all four projections share them)
#define GM 2048
#define GN 4096
#define GK 4096

__device__ inline unsigned short f2bf(float f) {
    union { float f; unsigned u; } v; v.f = f;
    unsigned r = v.u + 0x7FFFu + ((v.u >> 16) & 1u);
    return (unsigned short)(r >> 16);
}
__device__ inline float gelu_f(float x) {
    return 0.5f * x * (1.0f + erff(x * 0.7071067811865475f));
}
__device__ inline v8f vzero8() {
    v8f z;
#pragma unroll
    for (int i = 0; i < 8; ++i) z[i] = 0.f;
    return z;
}
__device__ inline v8f wmma_bf16(v16bf a, v16bf b, v8f c) {
    return __builtin_amdgcn_wmma_f32_16x16x32_bf16(false, a, false, b, (short)0, c, false, false);
}
// A-fragment (16x32, row-major source, ld elements): lane r=l&15; k in [k0+8h,+8) and [k0+16+8h,+8)
__device__ inline v16bf a_frag(const unsigned short* base, int ld, int r0, int k0, int lane) {
    int r = r0 + (lane & 15), hh = (lane >> 4) & 1;
    const unsigned short* p = base + (size_t)r * ld + k0 + 8 * hh;
    Frag u;
    u.q[0] = *reinterpret_cast<const q16*>(p);
    u.q[1] = *reinterpret_cast<const q16*>(p + 16);
    return u.v;
}
// B-fragment (32x16): element (k,n) at base[n*ld + k]; lane n=n0+(l&15); k in [k0+8h,+8) and [k0+16+8h,+8)
__device__ inline v16bf b_frag(const unsigned short* base, int ld, int n0, int k0, int lane) {
    int n = n0 + (lane & 15), hh = (lane >> 4) & 1;
    const unsigned short* p = base + (size_t)n * ld + k0 + 8 * hh;
    Frag u;
    u.q[0] = *reinterpret_cast<const q16*>(p);
    u.q[1] = *reinterpret_cast<const q16*>(p + 16);
    return u.v;
}

// ---------- bf16 WMMA GEMM: C(2048x4096) = A(2048x4096) @ B(4096x4096), f32 in/out ----------
// Block: 128 threads (4 waves), C tile 64x128; wave computes 32x64 (2x4 WMMAs).
__global__ __launch_bounds__(128) void gemm_bf16(const float* __restrict__ A,
                                                 const float* __restrict__ B,
                                                 float* __restrict__ C) {
    __shared__ __align__(16) unsigned short As[64 * 32];
    __shared__ __align__(16) unsigned short Bs[128 * 32]; // transposed: Bs[n*32 + k]
    const int m0 = blockIdx.y * 64, n0 = blockIdx.x * 128;
    const int t = threadIdx.x, lane = t & 31, w = t >> 5;
    const int wm = (w & 1) * 32, wn = (w >> 1) * 64;
    const int l15 = lane & 15, hh = (lane >> 4) & 1;
    v8f acc[2][4];
#pragma unroll
    for (int i = 0; i < 2; ++i)
#pragma unroll
        for (int j = 0; j < 4; ++j) acc[i][j] = vzero8();

    for (int k0 = 0; k0 < GK; k0 += 32) {
#pragma unroll
        for (int j = 0; j < 4; ++j) { // A tile 64x32
            int idx = t + 128 * j, r = idx >> 3, c = (idx & 7) * 4;
            float4 v = *reinterpret_cast<const float4*>(A + (size_t)(m0 + r) * GK + k0 + c);
            As[r * 32 + c + 0] = f2bf(v.x); As[r * 32 + c + 1] = f2bf(v.y);
            As[r * 32 + c + 2] = f2bf(v.z); As[r * 32 + c + 3] = f2bf(v.w);
        }
#pragma unroll
        for (int j = 0; j < 8; ++j) { // B tile 32x128, store transposed
            int idx = t + 128 * j, r = idx >> 5, c = (idx & 31) * 4;
            float4 v = *reinterpret_cast<const float4*>(B + (size_t)(k0 + r) * GN + n0 + c);
            Bs[(c + 0) * 32 + r] = f2bf(v.x); Bs[(c + 1) * 32 + r] = f2bf(v.y);
            Bs[(c + 2) * 32 + r] = f2bf(v.z); Bs[(c + 3) * 32 + r] = f2bf(v.w);
        }
        if (k0 + 32 < GK) { // prefetch next K tile (global_prefetch_b8)
            __builtin_prefetch(A + (size_t)(m0 + (t >> 3)) * GK + k0 + 32 + (t & 7) * 4, 0, 3);
            __builtin_prefetch(B + (size_t)(k0 + 32 + (t >> 5)) * GN + n0 + (t & 31) * 4, 0, 3);
        }
        __syncthreads();
        v16bf a0 = a_frag(As, 32, wm, 0, lane);
        v16bf a1 = a_frag(As, 32, wm + 16, 0, lane);
        v16bf b0 = b_frag(Bs, 32, wn, 0, lane);
        v16bf b1 = b_frag(Bs, 32, wn + 16, 0, lane);
        v16bf b2 = b_frag(Bs, 32, wn + 32, 0, lane);
        v16bf b3 = b_frag(Bs, 32, wn + 48, 0, lane);
        acc[0][0] = wmma_bf16(a0, b0, acc[0][0]);
        acc[0][1] = wmma_bf16(a0, b1, acc[0][1]);
        acc[0][2] = wmma_bf16(a0, b2, acc[0][2]);
        acc[0][3] = wmma_bf16(a0, b3, acc[0][3]);
        acc[1][0] = wmma_bf16(a1, b0, acc[1][0]);
        acc[1][1] = wmma_bf16(a1, b1, acc[1][1]);
        acc[1][2] = wmma_bf16(a1, b2, acc[1][2]);
        acc[1][3] = wmma_bf16(a1, b3, acc[1][3]);
        __syncthreads();
    }
    float* crow = C + (size_t)(m0 + wm + 8 * hh) * GN + n0 + wn + l15;
#pragma unroll
    for (int i = 0; i < 2; ++i)
#pragma unroll
        for (int j = 0; j < 4; ++j)
#pragma unroll
            for (int p = 0; p < 8; ++p)
                crow[(size_t)(16 * i + p) * GN + 16 * j] = acc[i][j][p];
}

// ---------- rotary + repack: q32/k32 rotated in place, bf16 q,k (h,s,hd) and vT (h,hd,s) ----------
__global__ __launch_bounds__(64) void rotary_pack(float* __restrict__ q32, float* __restrict__ k32,
                                                  const float* __restrict__ v32,
                                                  unsigned short* __restrict__ qb,
                                                  unsigned short* __restrict__ kb,
                                                  unsigned short* __restrict__ vT) {
    const int s = blockIdx.x, h = blockIdx.y, i = threadIdx.x; // i in [0,64)
    const float invf = powf(10000.f, -(float)i / 64.f);
    const float ang = (float)s * invf;
    const float c = cosf(ang), sn = sinf(ang);
    const size_t rowi = (size_t)s * D_DIM + h * HD_DIM;
    const size_t hb = (size_t)h * S_LEN * HD_DIM + (size_t)s * HD_DIM;

    float x1 = q32[rowi + i], x2 = q32[rowi + i + 64];
    float o1 = x1 * c - x2 * sn, o2 = x2 * c + x1 * sn;
    q32[rowi + i] = o1; q32[rowi + i + 64] = o2;
    qb[hb + i] = f2bf(o1); qb[hb + i + 64] = f2bf(o2);

    x1 = k32[rowi + i]; x2 = k32[rowi + i + 64];
    o1 = x1 * c - x2 * sn; o2 = x2 * c + x1 * sn;
    k32[rowi + i] = o1; k32[rowi + i + 64] = o2;
    kb[hb + i] = f2bf(o1); kb[hb + i + 64] = f2bf(o2);

    float va = v32[rowi + i], vb2 = v32[rowi + i + 64];
    vT[(size_t)(h * HD_DIM + i) * S_LEN + s] = f2bf(va);
    vT[(size_t)(h * HD_DIM + i + 64) * S_LEN + s] = f2bf(vb2);
}

// ---------- attention pass A: per-row softmax stats via WMMA q@k^T ----------
__global__ __launch_bounds__(128) void attn_stats(const unsigned short* __restrict__ qb,
                                                  const unsigned short* __restrict__ kb,
                                                  float* __restrict__ mrow, float* __restrict__ lrow) {
    const int h = blockIdx.y, q0 = blockIdx.x * 64;
    const int t = threadIdx.x, lane = t & 31, w = t >> 5;
    const int qr0 = q0 + 16 * w, l15 = lane & 15, hh = (lane >> 4) & 1;
    const unsigned short* qbase = qb + (size_t)h * S_LEN * HD_DIM;
    const unsigned short* kbase = kb + (size_t)h * S_LEN * HD_DIM;

    v16bf qa[4];
#pragma unroll
    for (int kc = 0; kc < 4; ++kc) qa[kc] = a_frag(qbase, HD_DIM, qr0, kc * 32, lane);

    float m[8], l[8];
#pragma unroll
    for (int p = 0; p < 8; ++p) { m[p] = -3.4e38f; l[p] = 0.f; }

    const int nch = (qr0 + 16 + 31) >> 5;
    for (int ch = 0; ch < nch; ++ch) {
        int k0 = ch * 32;
        v8f s0 = vzero8(), s1 = vzero8();
#pragma unroll
        for (int kc = 0; kc < 4; ++kc) {
            v16bf b0 = b_frag(kbase, HD_DIM, k0, kc * 32, lane);
            v16bf b1 = b_frag(kbase, HD_DIM, k0 + 16, kc * 32, lane);
            s0 = wmma_bf16(qa[kc], b0, s0);
            s1 = wmma_bf16(qa[kc], b1, s1);
        }
#pragma unroll
        for (int p = 0; p < 8; ++p) {
            int row = qr0 + 8 * hh + p;
            int c0 = k0 + l15, c1 = c0 + 16;
            float v0 = s0[p] * ATT_SCALE, v1 = s1[p] * ATT_SCALE;
            bool ok0 = (c0 <= row), ok1 = (c1 <= row);
            float cm = fmaxf(ok0 ? v0 : -3.4e38f, ok1 ? v1 : -3.4e38f);
#pragma unroll
            for (int d = 8; d >= 1; d >>= 1) cm = fmaxf(cm, __shfl_xor(cm, d, 32));
            float nm = fmaxf(m[p], cm);
            float e0 = ok0 ? expf(v0 - nm) : 0.f;
            float e1 = ok1 ? expf(v1 - nm) : 0.f;
            float ps = e0 + e1;
#pragma unroll
            for (int d = 8; d >= 1; d >>= 1) ps += __shfl_xor(ps, d, 32);
            l[p] = l[p] * expf(m[p] - nm) + ps;
            m[p] = nm;
        }
    }
#pragma unroll
    for (int p = 0; p < 8; ++p) {
        if (l15 == p) {
            int row = qr0 + 8 * hh + p;
            mrow[(size_t)h * S_LEN + row] = m[p];
            lrow[(size_t)h * S_LEN + row] = l[p];
        }
    }
}

// ---------- attention pass B: exact P, O = P@V (WMMA), scores += colsum(P) ----------
// V tile (128 hd x 32 keys bf16, 8KB) is fetched once per chunk for the whole block
// with GLOBAL_LOAD_ASYNC_TO_LDS_B128 (ASYNCcnt), then consumed by all 4 waves.
__global__ __launch_bounds__(128) void attn_av(const unsigned short* __restrict__ qb,
                                               const unsigned short* __restrict__ kb,
                                               const unsigned short* __restrict__ vT,
                                               const float* __restrict__ mrow,
                                               const float* __restrict__ lrow,
                                               float* __restrict__ scores,
                                               float* __restrict__ attn_out) {
    __shared__ __align__(16) unsigned short Pb[4][16][32]; // per-wave P tile (bf16)
    __shared__ __align__(16) unsigned short Vs[128 * 32];  // V tile: Vs[hd*32 + klocal]
    const int h = blockIdx.y, q0 = blockIdx.x * 64;
    const int t = threadIdx.x, lane = t & 31, w = t >> 5;
    const int qr0 = q0 + 16 * w, l15 = lane & 15, hh = (lane >> 4) & 1;
    const unsigned short* qbase = qb + (size_t)h * S_LEN * HD_DIM;
    const unsigned short* kbase = kb + (size_t)h * S_LEN * HD_DIM;
    const unsigned short* vbase = vT + (size_t)h * HD_DIM * S_LEN;

    // LDS aperture: low 32 bits of a flat shared pointer are the LDS byte offset (ISA 10.2)
    const unsigned vs_lds = (unsigned)(size_t)(&Vs[0]) + (unsigned)(t * 64);
    const unsigned long long vgbase = (unsigned long long)(size_t)vbase;

    v16bf qa[4];
#pragma unroll
    for (int kc = 0; kc < 4; ++kc) qa[kc] = a_frag(qbase, HD_DIM, qr0, kc * 32, lane);

    float mr[8], rl[8];
#pragma unroll
    for (int p = 0; p < 8; ++p) {
        int row = qr0 + 8 * hh + p;
        mr[p] = mrow[(size_t)h * S_LEN + row];
        rl[p] = 1.0f / lrow[(size_t)h * S_LEN + row];
    }
    v8f O[8];
#pragma unroll
    for (int j = 0; j < 8; ++j) O[j] = vzero8();

    const int nch = (q0 >> 5) + 2; // uniform across the block (covers rows q0..q0+63)
    for (int ch = 0; ch < nch; ++ch) {
        int k0 = ch * 32;
        // async copy of V tile: thread t moves hd-row t (64 bytes) in 4 x b128
        {
            unsigned go = (unsigned)((t * S_LEN + k0) * 2);
#pragma unroll
            for (int ii = 0; ii < 4; ++ii) {
                asm volatile("global_load_async_to_lds_b128 %0, %1, %2"
                             :: "v"(vs_lds + 16 * ii), "v"(go + 16 * ii), "s"(vgbase)
                             : "memory");
            }
        }
        v8f s0 = vzero8(), s1 = vzero8();
#pragma unroll
        for (int kc = 0; kc < 4; ++kc) {
            v16bf b0 = b_frag(kbase, HD_DIM, k0, kc * 32, lane);
            v16bf b1 = b_frag(kbase, HD_DIM, k0 + 16, kc * 32, lane);
            s0 = wmma_bf16(qa[kc], b0, s0);
            s1 = wmma_bf16(qa[kc], b1, s1);
        }
        float cs0 = 0.f, cs1 = 0.f;
#pragma unroll
        for (int p = 0; p < 8; ++p) {
            int row = qr0 + 8 * hh + p;
            int c0 = k0 + l15, c1 = c0 + 16;
            float p0 = (c0 <= row) ? expf(s0[p] * ATT_SCALE - mr[p]) * rl[p] : 0.f;
            float p1 = (c1 <= row) ? expf(s1[p] * ATT_SCALE - mr[p]) * rl[p] : 0.f;
            cs0 += p0; cs1 += p1;
            Pb[w][8 * hh + p][l15] = f2bf(p0);
            Pb[w][8 * hh + p][16 + l15] = f2bf(p1);
        }
        atomicAdd(scores + (size_t)h * S_LEN + k0 + l15, cs0);
        atomicAdd(scores + (size_t)h * S_LEN + k0 + 16 + l15, cs1);
        asm volatile("s_wait_asynccnt 0" ::: "memory");
        __syncthreads();
        v16bf pa = a_frag(&Pb[w][0][0], 32, 0, 0, lane);
#pragma unroll
        for (int j2 = 0; j2 < 8; ++j2) {
            v16bf bv = b_frag(Vs, 32, 16 * j2, 0, lane);
            O[j2] = wmma_bf16(pa, bv, O[j2]);
        }
        __syncthreads();
    }
#pragma unroll
    for (int j2 = 0; j2 < 8; ++j2)
#pragma unroll
        for (int p = 0; p < 8; ++p)
            attn_out[(size_t)(qr0 + 8 * hh + p) * D_DIM + h * HD_DIM + 16 * j2 + l15] = O[j2][p];
}

// ---------- zero init ----------
__global__ void zero_f32(float* p, int n) {
    int i = blockIdx.x * blockDim.x + threadIdx.x;
    if (i < n) p[i] = 0.f;
}

// ---------- top-k heavy-hitter mask -> elim ----------
__global__ __launch_bounds__(256) void topk_mask(const float* __restrict__ scores,
                                                 float* __restrict__ elim) {
    __shared__ float sc[S_LEN - RECENT_N];   // 1792
    __shared__ float rv[256];
    __shared__ int   ri[256];
    const int h = blockIdx.x, t = threadIdx.x;
    const int NSEL = S_LEN - RECENT_N; // 1792
    for (int i = t; i < NSEL; i += 256) sc[i] = scores[(size_t)h * S_LEN + i];
    for (int i = t; i < S_LEN; i += 256)
        elim[(size_t)h * S_LEN + i] = (i < (S_LEN + 1 - RECENT_N)) ? 1.f : 0.f;
    __syncthreads();
    for (int iter = 0; iter < HEAVY_N; ++iter) {
        float lm = -3.4e38f; int li = 0;
        for (int i = t; i < NSEL; i += 256)
            if (sc[i] > lm) { lm = sc[i]; li = i; }
        rv[t] = lm; ri[t] = li;
        __syncthreads();
        for (int off = 128; off > 0; off >>= 1) {
            if (t < off && rv[t + off] > rv[t]) { rv[t] = rv[t + off]; ri[t] = ri[t + off]; }
            __syncthreads();
        }
        if (t == 0) {
            int wi = ri[0];
            sc[wi] = -3.4e38f;
            elim[(size_t)h * S_LEN + wi] = 0.f;
        }
        __syncthreads();
    }
}

// ---------- q_ker chain: |gelu(gelu(q@kq1)@kq2)| ----------
__global__ __launch_bounds__(128) void qker_kernel(const float* __restrict__ q32,
                                                   const float* __restrict__ kq1,
                                                   const float* __restrict__ kq2,
                                                   float* __restrict__ qker) {
    __shared__ float X[16 * 128];
    __shared__ float T1[16 * 128];
    const int h = blockIdx.y, s0 = blockIdx.x * 16, t = threadIdx.x;
#pragma unroll
    for (int j = 0; j < 16; ++j) {
        int idx = t + 128 * j, r = idx >> 7, d = idx & 127;
        X[idx] = q32[(size_t)(s0 + r) * D_DIM + h * HD_DIM + d];
    }
    __syncthreads();
    const float* W1 = kq1 + (size_t)h * HD_DIM * KH_DIM;
#pragma unroll
    for (int j = 0; j < 16; ++j) {
        int o = t + 128 * j, r = o >> 7, f = o & 127;
        float s = 0.f;
        for (int d = 0; d < 128; ++d) s += X[r * 128 + d] * W1[d * 128 + f];
        T1[o] = gelu_f(s);
    }
    __syncthreads();
    const float* W2 = kq2 + (size_t)h * KH_DIM * KD_DIM;
#pragma unroll
    for (int j = 0; j < 8; ++j) {
        int o = t + 128 * j, r = o >> 6, f = o & 63;
        float s = 0.f;
        for (int e = 0; e < 128; ++e) s += T1[r * 128 + e] * W2[e * 64 + f];
        qker[(size_t)h * S_LEN * KD_DIM + (size_t)(s0 + r) * KD_DIM + f] = fabsf(gelu_f(s));
    }
}

// ---------- k_ker chain with interaction ----------
__global__ __launch_bounds__(128) void kker_kernel(const float* __restrict__ k32,
                                                   const float* __restrict__ kk1,
                                                   const float* __restrict__ kk2,
                                                   const float* __restrict__ scalingD,
                                                   const float* __restrict__ ik,
                                                   const float* __restrict__ scalingD2,
                                                   float* __restrict__ kker) {
    __shared__ float X[16 * 128];
    __shared__ float T1[16 * 128];
    __shared__ float T2[16 * 64];
    const int h = blockIdx.y, s0 = blockIdx.x * 16, t = threadIdx.x;
#pragma unroll
    for (int j = 0; j < 16; ++j) {
        int idx = t + 128 * j, r = idx >> 7, d = idx & 127;
        X[idx] = k32[(size_t)(s0 + r) * D_DIM + h * HD_DIM + d];
    }
    __syncthreads();
    const float* W1 = kk1 + (size_t)h * HD_DIM * KH_DIM;
#pragma unroll
    for (int j = 0; j < 16; ++j) {
        int o = t + 128 * j, r = o >> 7, f = o & 127;
        float s = 0.f;
        for (int d = 0; d < 128; ++d) s += X[r * 128 + d] * W1[d * 128 + f];
        T1[o] = gelu_f(s);
    }
    __syncthreads();
    const float* W2 = kk2 + (size_t)h * KH_DIM * KD_DIM;
#pragma unroll
    for (int j = 0; j < 8; ++j) {
        int o = t + 128 * j, r = o >> 6, f = o & 63;
        float s = 0.f;
        for (int e = 0; e < 128; ++e) s += T1[r * 128 + e] * W2[e * 64 + f];
        T2[o] = fabsf(scalingD[h * KD_DIM + f]) * gelu_f(s);
    }
    __syncthreads();
    const float* IK = ik + (size_t)h * KD_DIM * KD_DIM;
#pragma unroll
    for (int j = 0; j < 8; ++j) {
        int o = t + 128 * j, r = o >> 6, f = o & 63;
        float s3 = 0.f;
        for (int e = 0; e < 64; ++e) s3 += T2[r * 64 + e] * IK[e * 64 + f];
        float kv = fabsf(T2[r * 64 + f] + s3 * scalingD2[h * KD_DIM + f]);
        kker[(size_t)h * S_LEN * KD_DIM + (size_t)(s0 + r) * KD_DIM + f] = kv;
    }
}

// ---------- H_new / z_new ----------
__global__ __launch_bounds__(128) void hnew_kernel(const float* __restrict__ kker,
                                                   const float* __restrict__ elim,
                                                   const float* __restrict__ v32,
                                                   float* __restrict__ Hn,
                                                   float* __restrict__ zn) {
    const int e = blockIdx.x, h = blockIdx.y, d = threadIdx.x;
    float acc = 0.f, zacc = 0.f;
    for (int s = 0; s < S_LEN; ++s) {
        float wgt = elim[(size_t)h * S_LEN + s] * kker[(size_t)h * S_LEN * KD_DIM + (size_t)s * KD_DIM + e];
        acc += wgt * v32[(size_t)s * D_DIM + h * HD_DIM + d];
        zacc += wgt;
    }
    Hn[(size_t)h * KD_DIM * HD_DIM + (size_t)e * HD_DIM + d] = acc;
    if (d == 0) zn[(size_t)h * KD_DIM + e] = zacc;
}

// ---------- launch ----------
extern "C" void kernel_launch(void* const* d_in, const int* in_sizes, int n_in,
                              void* d_out, int out_size, void* d_ws, size_t ws_size,
                              hipStream_t stream) {
    const float* hs  = (const float*)d_in[0];
    const float* wq  = (const float*)d_in[1];
    const float* wk  = (const float*)d_in[2];
    const float* wv  = (const float*)d_in[3];
    const float* wo  = (const float*)d_in[4];
    const float* kq1 = (const float*)d_in[5];
    const float* kq2 = (const float*)d_in[6];
    const float* kk1 = (const float*)d_in[7];
    const float* kk2 = (const float*)d_in[8];
    const float* scD = (const float*)d_in[9];
    const float* ik  = (const float*)d_in[10];
    const float* scD2= (const float*)d_in[11];
    float* out = (float*)d_out;

    const size_t SD = (size_t)S_LEN * D_DIM;      // 8388608
    char* ws = (char*)d_ws;
    float* q32 = (float*)(ws);
    float* k32 = (float*)(ws + SD * 4);
    float* v32 = (float*)(ws + SD * 8);
    unsigned short* qb  = (unsigned short*)(ws + SD * 12);
    unsigned short* kb  = (unsigned short*)(ws + SD * 14);
    unsigned short* vTb = (unsigned short*)(ws + SD * 16);
    float* attn_out = (float*)(ws + SD * 18);
    float* kker  = (float*)(ws + SD * 22);                 // H*S*KD f32 = SD*2 bytes
    float* mrow  = (float*)(ws + SD * 24);
    float* lrow  = (float*)(ws + SD * 24 + 262144);
    float* scores= (float*)(ws + SD * 24 + 524288);
    float* elim  = (float*)(ws + SD * 24 + 786432);

    float* out_main = out;                         // (1,S,D)
    float* out_Hnew = out + SD;                    // (1,H,KD,HD)
    float* out_znew = out + SD + H_NUM * KD_DIM * HD_DIM;
    float* out_qker = out + SD + H_NUM * KD_DIM * HD_DIM + H_NUM * KD_DIM;

    dim3 gGemm(GN / 128, GM / 64);
    gemm_bf16<<<gGemm, 128, 0, stream>>>(hs, wq, q32);
    gemm_bf16<<<gGemm, 128, 0, stream>>>(hs, wk, k32);
    gemm_bf16<<<gGemm, 128, 0, stream>>>(hs, wv, v32);

    rotary_pack<<<dim3(S_LEN, H_NUM), 64, 0, stream>>>(q32, k32, v32, qb, kb, vTb);

    zero_f32<<<(H_NUM * S_LEN + 255) / 256, 256, 0, stream>>>(scores, H_NUM * S_LEN);

    attn_stats<<<dim3(S_LEN / 64, H_NUM), 128, 0, stream>>>(qb, kb, mrow, lrow);
    attn_av<<<dim3(S_LEN / 64, H_NUM), 128, 0, stream>>>(qb, kb, vTb, mrow, lrow, scores, attn_out);

    topk_mask<<<H_NUM, 256, 0, stream>>>(scores, elim);

    qker_kernel<<<dim3(S_LEN / 16, H_NUM), 128, 0, stream>>>(q32, kq1, kq2, out_qker);
    kker_kernel<<<dim3(S_LEN / 16, H_NUM), 128, 0, stream>>>(k32, kk1, kk2, scD, ik, scD2, kker);
    hnew_kernel<<<dim3(KD_DIM, H_NUM), 128, 0, stream>>>(kker, elim, v32, out_Hnew, out_znew);

    gemm_bf16<<<gGemm, 128, 0, stream>>>(attn_out, wo, out_main);
}
